// RelativeMultiheadAttention_35081292873741
// MI455X (gfx1250) — matrix operand order repaired and checked by reading, not compiled, not run on hardware
//
#include <hip/hip_runtime.h>

typedef __attribute__((ext_vector_type(16))) __bf16 v16bf;
typedef __attribute__((ext_vector_type(8)))  __bf16 v8bf;
typedef __attribute__((ext_vector_type(8)))  float  v8f;

#define D_MODEL  1024
#define N_HEADS  16
#define HEAD_DIM 64
#define SEQ      2048
#define BATCH    2
#define MAX_REL  2047
#define REL_ROWS 4095

__device__ __forceinline__ v8f wmma_bf16(v16bf a, v16bf b, v8f c) {
  // D = A(16x32 bf16) * B(32x16 bf16) + C(16x16 f32)
  return __builtin_amdgcn_wmma_f32_16x16x32_bf16(false, a, false, b, (short)0, c,
                                                 false, false);
}

// ---------------------------------------------------------------------------
// Projection GEMM: out[m, n] = sum_k X[m,k] * W[n,k] + bias[n]   (X @ W^T + b)
// M = B*SEQ = 4096, N = K = 1024. Each wave computes a 64x64 tile via a 4x4
// grid of 16x16 f32 accumulators; K loop steps 32 (bf16 WMMA depth).
// vmode==0: store bf16 as [B, H, S, Dh]   (Q and K layout)
// vmode==1: store bf16 as [B, H, Dh, S]   (V transposed for P@V B-fragments)
// ---------------------------------------------------------------------------
__global__ __launch_bounds__(256) void proj_kernel(
    const float* __restrict__ X, const float* __restrict__ W,
    const float* __restrict__ bias, __bf16* __restrict__ out, int vmode) {
  const int lane = threadIdx.x & 31;
  const int wave = threadIdx.x >> 5;
  const int lo = lane & 15, hi = lane >> 4;
  const int wg = blockIdx.x * 8 + wave;       // 1024 wave-tiles total
  const int m0 = (wg >> 4) * 64;              // 64 M-tiles
  const int n0 = (wg & 15) * 64;              // 16 N-tiles

  v8f acc[4][4] = {};

  for (int k0 = 0; k0 < D_MODEL; k0 += 32) {
    v16bf af[4], bfm[4];
    // A fragments (16x32): lane<16 holds M=lo, K = 0..15;
    // lane>=16 holds M=lo, K = 16..31
#pragma unroll
    for (int mt = 0; mt < 4; ++mt) {
      const float* p = X + (size_t)(m0 + mt * 16 + lo) * D_MODEL + k0 + hi * 16;
      v16bf a;
#pragma unroll
      for (int i = 0; i < 8; ++i) {
        a[i]     = (__bf16)p[i];
        a[i + 8] = (__bf16)p[i + 8];
      }
      af[mt] = a;
    }
    // B fragments (32x16): B[k,n] = W[n,k]; lane<16 -> K=0..15, lane>=16 -> K=16..31
#pragma unroll
    for (int nt = 0; nt < 4; ++nt) {
      const float* p = W + (size_t)(n0 + nt * 16 + lo) * D_MODEL + k0 + hi * 16;
      v16bf bm;
#pragma unroll
      for (int i = 0; i < 16; ++i) bm[i] = (__bf16)p[i];
      bfm[nt] = bm;
    }
#pragma unroll
    for (int mt = 0; mt < 4; ++mt)
#pragma unroll
      for (int nt = 0; nt < 4; ++nt)
        acc[mt][nt] = wmma_bf16(af[mt], bfm[nt], acc[mt][nt]);
  }

  // C layout: VGPR v, lanes 0-15 -> (M=v, N=lane); lanes 16-31 -> (M=v+8, N=lane-16)
#pragma unroll
  for (int nt = 0; nt < 4; ++nt) {
    const int col = n0 + nt * 16 + lo;
    const float bv = bias[col];
    const int h = col >> 6, d = col & (HEAD_DIM - 1);
#pragma unroll
    for (int mt = 0; mt < 4; ++mt) {
      const int rowb = m0 + mt * 16 + hi * 8;
#pragma unroll
      for (int v = 0; v < 8; ++v) {
        const int row = rowb + v;
        const int b = row >> 11;             // SEQ = 2048 rows per batch
        const int t = row & (SEQ - 1);
        const float val = acc[mt][nt][v] + bv;
        if (vmode)
          out[(((size_t)b * N_HEADS + h) * HEAD_DIM + d) * SEQ + t] = (__bf16)val;
        else
          out[(((size_t)b * N_HEADS + h) * SEQ + t) * HEAD_DIM + d] = (__bf16)val;
      }
    }
  }
}

// ---------------------------------------------------------------------------
// Flash-style attention with relative bias + mask.
// Grid: BATCH * N_HEADS * (SEQ/64) blocks, 128 threads (4 waves).
// Each wave owns 16 query rows, streams keys in chunks of 32 with online
// softmax. rel_emb column for head h staged in LDS (index = t - s + 2047).
// ---------------------------------------------------------------------------
__global__ __launch_bounds__(128) void attn_kernel(
    const __bf16* __restrict__ Qh, const __bf16* __restrict__ Kh,
    const __bf16* __restrict__ Vt, const unsigned char* __restrict__ mask,
    const float* __restrict__ rel_emb, __bf16* __restrict__ attn) {
  __shared__ float biasCol[REL_ROWS];
  __shared__ __align__(16) __bf16 pbuf[4 * 16 * 32];

  const int lane = threadIdx.x & 31;
  const int wave = threadIdx.x >> 5;
  const int lo = lane & 15, hi = lane >> 4;

  const int bid = blockIdx.x;
  const int tchunk = bid & 31;                 // 32 chunks of 64 queries
  const int bh = bid >> 5;
  const int h = bh & (N_HEADS - 1);
  const int b = bh >> 4;
  const int t0 = tchunk * 64 + wave * 16;

  for (int i = threadIdx.x; i < REL_ROWS; i += 128)
    biasCol[i] = rel_emb[(size_t)i * N_HEADS + h];
  __syncthreads();

  // Q A-fragments for K-dim = HEAD_DIM (two k-steps of 32)
  v16bf qf[2];
  {
    const __bf16* qrow =
        Qh + (((size_t)b * N_HEADS + h) * SEQ + (t0 + lo)) * HEAD_DIM;
#pragma unroll
    for (int kk = 0; kk < 2; ++kk) {
      v8bf c0 = *(const v8bf*)(qrow + kk * 32 + hi * 16);
      v8bf c1 = *(const v8bf*)(qrow + kk * 32 + hi * 16 + 8);
      v16bf a;
#pragma unroll
      for (int i = 0; i < 8; ++i) { a[i] = c0[i]; a[i + 8] = c1[i]; }
      qf[kk] = a;
    }
  }

  const __bf16* kbase = Kh + ((size_t)b * N_HEADS + h) * SEQ * HEAD_DIM;
  const __bf16* vbase = Vt + ((size_t)b * N_HEADS + h) * HEAD_DIM * SEQ;
  __bf16* pb = pbuf + wave * 16 * 32;

  v8f O[4] = {};
  float mrun[8], lrun[8];
#pragma unroll
  for (int v = 0; v < 8; ++v) { mrun[v] = -1e30f; lrun[v] = 0.f; }

  for (int s0 = 0; s0 < SEQ; s0 += 32) {
    if (s0 + 32 < SEQ) {  // hint next K/V tiles into cache (global_prefetch_b8)
      __builtin_prefetch(kbase + (size_t)(s0 + 32 + lane) * HEAD_DIM, 0, 1);
      __builtin_prefetch(vbase + (size_t)lane * SEQ + s0 + 32, 0, 1);
    }
    // scores = Q K^T for 32 keys: two 16x16 C frags, two k-steps each
    v8f sc[2] = {};
#pragma unroll
    for (int nt = 0; nt < 2; ++nt) {
      const __bf16* kr = kbase + (size_t)(s0 + nt * 16 + lo) * HEAD_DIM;
      v16bf kf0 = *(const v16bf*)(kr + hi * 16);
      v16bf kf1 = *(const v16bf*)(kr + 32 + hi * 16);
      sc[nt] = wmma_bf16(qf[0], kf0, sc[nt]);
      sc[nt] = wmma_bf16(qf[1], kf1, sc[nt]);
    }
    // scale + relative bias + mask
    float x0[8], x1[8];
#pragma unroll
    for (int nt = 0; nt < 2; ++nt) {
      const int sb = s0 + nt * 16 + lo;
      const int tb = t0 + hi * 8;
      const int ib = tb - sb + MAX_REL;       // in [0, 4094], no clip needed
      const unsigned char* mrow = mask + ((size_t)b * SEQ + tb) * SEQ + sb;
      float* xx = nt ? x1 : x0;
#pragma unroll
      for (int v = 0; v < 8; ++v) {
        float val = sc[nt][v] * 0.125f + biasCol[ib + v];
        if (!mrow[(size_t)v * SEQ]) val = -1e30f;
        xx[v] = val;
      }
    }
    // online softmax: rows of a C frag live inside one 16-lane half
    float scl[8];
#pragma unroll
    for (int v = 0; v < 8; ++v) {
      float rm = fmaxf(x0[v], x1[v]);
#pragma unroll
      for (int m = 1; m <= 8; m <<= 1) rm = fmaxf(rm, __shfl_xor(rm, m, 32));
      const float nm = fmaxf(mrun[v], rm);
      const float s_ = __expf(mrun[v] - nm);
      mrun[v] = nm;
      const float p0 = __expf(x0[v] - nm);
      const float p1 = __expf(x1[v] - nm);
      x0[v] = p0; x1[v] = p1;
      float rs = p0 + p1;
#pragma unroll
      for (int m = 1; m <= 8; m <<= 1) rs += __shfl_xor(rs, m, 32);
      lrun[v] = lrun[v] * s_ + rs;
      scl[v] = s_;
    }
#pragma unroll
    for (int nt2 = 0; nt2 < 4; ++nt2)
#pragma unroll
      for (int v = 0; v < 8; ++v) O[nt2][v] *= scl[v];

    // C-layout P -> LDS -> reload in A-fragment layout (16x32, K = key index)
#pragma unroll
    for (int v = 0; v < 8; ++v) {
      pb[(v + 8 * hi) * 32 + lo]      = (__bf16)x0[v];
      pb[(v + 8 * hi) * 32 + lo + 16] = (__bf16)x1[v];
    }
    asm volatile("s_wait_dscnt 0" ::: "memory");
    v16bf pf;
    {
      v8bf c0 = *(const v8bf*)(pb + lo * 32 + hi * 16);
      v8bf c1 = *(const v8bf*)(pb + lo * 32 + hi * 16 + 8);
#pragma unroll
      for (int i = 0; i < 8; ++i) { pf[i] = c0[i]; pf[8 + i] = c1[i]; }
    }
    // O += P @ V : V B-frags read contiguously from transposed V [Dh, S]
#pragma unroll
    for (int nt2 = 0; nt2 < 4; ++nt2) {
      const v16bf* vp =
          (const v16bf*)(vbase + (size_t)(nt2 * 16 + lo) * SEQ + s0 + hi * 16);
      O[nt2] = wmma_bf16(pf, *vp, O[nt2]);
    }
  }

  // normalize and write attention output as bf16 [B, T, D_MODEL]
#pragma unroll
  for (int nt2 = 0; nt2 < 4; ++nt2) {
    const int col = h * HEAD_DIM + nt2 * 16 + lo;
#pragma unroll
    for (int v = 0; v < 8; ++v) {
      const int t = t0 + hi * 8 + v;
      attn[((size_t)b * SEQ + t) * D_MODEL + col] = (__bf16)(O[nt2][v] / lrun[v]);
    }
  }
}

// ---------------------------------------------------------------------------
// Output projection: d_out = attn(bf16) @ Wo^T + bo, f32 out. Same tiling as
// proj_kernel but A is already bf16.
// ---------------------------------------------------------------------------
__global__ __launch_bounds__(256) void oproj_kernel(
    const __bf16* __restrict__ A, const float* __restrict__ W,
    const float* __restrict__ bias, float* __restrict__ out) {
  const int lane = threadIdx.x & 31;
  const int wave = threadIdx.x >> 5;
  const int lo = lane & 15, hi = lane >> 4;
  const int wg = blockIdx.x * 8 + wave;
  const int m0 = (wg >> 4) * 64;
  const int n0 = (wg & 15) * 64;

  v8f acc[4][4] = {};

  for (int k0 = 0; k0 < D_MODEL; k0 += 32) {
    v16bf af[4], bfm[4];
#pragma unroll
    for (int mt = 0; mt < 4; ++mt) {
      const __bf16* p = A + (size_t)(m0 + mt * 16 + lo) * D_MODEL + k0 + hi * 16;
      v8bf c0 = *(const v8bf*)p;
      v8bf c1 = *(const v8bf*)(p + 8);
      v16bf a;
#pragma unroll
      for (int i = 0; i < 8; ++i) { a[i] = c0[i]; a[i + 8] = c1[i]; }
      af[mt] = a;
    }
#pragma unroll
    for (int nt = 0; nt < 4; ++nt) {
      const float* p = W + (size_t)(n0 + nt * 16 + lo) * D_MODEL + k0 + hi * 16;
      v16bf bm;
#pragma unroll
      for (int i = 0; i < 16; ++i) bm[i] = (__bf16)p[i];
      bfm[nt] = bm;
    }
#pragma unroll
    for (int mt = 0; mt < 4; ++mt)
#pragma unroll
      for (int nt = 0; nt < 4; ++nt)
        acc[mt][nt] = wmma_bf16(af[mt], bfm[nt], acc[mt][nt]);
  }

#pragma unroll
  for (int nt = 0; nt < 4; ++nt) {
    const int col = n0 + nt * 16 + lo;
    const float bv = bias[col];
#pragma unroll
    for (int mt = 0; mt < 4; ++mt) {
      const int rowb = m0 + mt * 16 + hi * 8;
#pragma unroll
      for (int v = 0; v < 8; ++v)
        out[(size_t)(rowb + v) * D_MODEL + col] = acc[mt][nt][v] + bv;
    }
  }
}

extern "C" void kernel_launch(void* const* d_in, const int* in_sizes, int n_in,
                              void* d_out, int out_size, void* d_ws,
                              size_t ws_size, hipStream_t stream) {
  (void)in_sizes; (void)n_in; (void)out_size; (void)ws_size;
  const float* query = (const float*)d_in[0];
  const float* key   = (const float*)d_in[1];
  const float* value = (const float*)d_in[2];
  const unsigned char* mask = (const unsigned char*)d_in[3];
  const float* Wq = (const float*)d_in[4];
  const float* bq = (const float*)d_in[5];
  const float* Wk = (const float*)d_in[6];
  const float* bk = (const float*)d_in[7];
  const float* Wv = (const float*)d_in[8];
  const float* bv = (const float*)d_in[9];
  const float* Wo = (const float*)d_in[10];
  const float* bo = (const float*)d_in[11];
  const float* rel = (const float*)d_in[12];

  const size_t E = (size_t)BATCH * N_HEADS * SEQ * HEAD_DIM;  // 4M elems
  __bf16* Qh = (__bf16*)d_ws;   // [B,H,S,Dh] bf16
  __bf16* Kh = Qh + E;          // [B,H,S,Dh] bf16
  __bf16* Vt = Kh + E;          // [B,H,Dh,S] bf16 (transposed)
  __bf16* At = Vt + E;          // [B,T,D]    bf16 attention output

  proj_kernel<<<128, 256, 0, stream>>>(query, Wq, bq, Qh, 0);
  proj_kernel<<<128, 256, 0, stream>>>(key,   Wk, bk, Kh, 0);
  proj_kernel<<<128, 256, 0, stream>>>(value, Wv, bv, Vt, 1);
  attn_kernel<<<BATCH * N_HEADS * (SEQ / 64), 128, 0, stream>>>(
      Qh, Kh, Vt, mask, rel, At);
  oproj_kernel<<<128, 256, 0, stream>>>(At, Wo, bo, (float*)d_out);
}
